// EncoderLayer_74990128988660
// MI455X (gfx1250) — matrix operand order repaired and checked by reading, not compiled, not run on hardware
//
#include <hip/hip_runtime.h>
#include <hip/hip_bf16.h>
#include <cstdint>

typedef unsigned int u32;
typedef unsigned short u16;
typedef __attribute__((ext_vector_type(16))) __bf16 v16bf;
typedef __attribute__((ext_vector_type(8)))  float  v8f;

union FragB { v16bf v; u32 u[8]; u16 s[16]; };
union FragC { v8f  v; float f[8]; };

__device__ __forceinline__ u16 f2bf(float f) {
  u32 u = __float_as_uint(f);
  u32 r = u + 0x7FFFu + ((u >> 16) & 1u);   // round-to-nearest-even
  return (u16)(r >> 16);
}

// ---- CDNA5 async global->LDS staging (guarded: falls back to VGPR staging) ----
#if defined(__AMDGCN__) && __has_builtin(__builtin_amdgcn_global_load_async_to_lds_b128)
#define USE_ASYNC_LDS 1
#else
#define USE_ASYNC_LDS 0
#endif

#if USE_ASYNC_LDS
typedef int v4i __attribute__((vector_size(16)));   // matches builtin's generic vector type
__device__ __forceinline__ void async_cp16(u16* lds, const u16* g) {
  __builtin_amdgcn_global_load_async_to_lds_b128(
      (__attribute__((address_space(1))) v4i*)(uintptr_t)g,
      (__attribute__((address_space(3))) v4i*)(unsigned)(uintptr_t)lds,
      0, 0);
}
__device__ __forceinline__ void wait_async0() {
#if __has_builtin(__builtin_amdgcn_s_wait_asynccnt)
  __builtin_amdgcn_s_wait_asynccnt(0);
#else
  asm volatile("s_wait_asynccnt 0" ::: "memory");
#endif
}
#endif

static constexpr int Bb = 4, S = 2048, D = 1024, H = 16, DK = 64, DFF = 2048;
static constexpr int BS = Bb * S;   // 8192 rows

// ---------------- conversion / packing ----------------
__global__ void k_cvt_bf16(const float* __restrict__ src, u16* __restrict__ dst, int n) {
  int i = blockIdx.x * blockDim.x + threadIdx.x;
  if (i < n) dst[i] = f2bf(src[i]);
}

// wq[h][d][k]  ->  W'[d][h*64+k]  (bf16), so QKV proj is one 1024x1024 GEMM
__global__ void k_pack_qkvw(const float* __restrict__ src, u16* __restrict__ dst) {
  int i = blockIdx.x * blockDim.x + threadIdx.x;   // over D*D
  int d = i >> 10; int n = i & 1023; int h = n >> 6; int kk = n & 63;
  dst[i] = f2bf(src[(size_t)(h * D + d) * DK + kk]);
}

// ---------------- bf16 WMMA GEMM: C = A(MxK) * B(KxN) (+bias)(+relu)(+resid) ----------------
__global__ __launch_bounds__(256) void k_gemm(
    const u16* __restrict__ A, const u16* __restrict__ Bm,
    const float* __restrict__ bias, const float* __restrict__ resid,
    float* __restrict__ outF, u16* __restrict__ outB,
    int M, int N, int K, int relu)
{
  __shared__ __align__(16) u16 ldsA[128 * 40];   // [row][k], stride 40 (80B, 16B aligned)
  __shared__ __align__(16) u16 ldsB[128 * 40];   // transposed: [n][k]

  const int t    = threadIdx.x;
  const int lane = t & 31;
  const int hi   = lane >> 4;
  const int ln   = lane & 15;
  const int wave = t >> 5;
  const int waveM = wave >> 1;        // 0..3 -> 32-row strip
  const int waveN = wave & 1;         // 0..1 -> 64-col strip
  const int mBase = blockIdx.y * 128;
  const int nBase = blockIdx.x * 128;

  FragC acc[2][4];
  #pragma unroll
  for (int mi = 0; mi < 2; ++mi)
    #pragma unroll
    for (int ni = 0; ni < 4; ++ni)
      #pragma unroll
      for (int r = 0; r < 8; ++r) acc[mi][ni].f[r] = 0.f;

  const int aRow  = t >> 1;            // 0..127
  const int aKoff = (t & 1) * 16;      // 0 / 16
  const int bK    = t >> 3;            // 0..31
  const int bN    = (t & 7) * 16;      // 0..112

  for (int kb = 0; kb < K; kb += 32) {
    // stage A tile (row-major): async DMA to LDS when available
    {
      const u16* gp = A + (size_t)(mBase + aRow) * K + kb + aKoff;
#if USE_ASYNC_LDS
      async_cp16(&ldsA[aRow * 40 + aKoff],     gp);
      async_cp16(&ldsA[aRow * 40 + aKoff + 8], gp + 8);
#else
      uint4 v0 = *(const uint4*)gp;
      uint4 v1 = *(const uint4*)(gp + 8);
      *(uint4*)&ldsA[aRow * 40 + aKoff]     = v0;
      *(uint4*)&ldsA[aRow * 40 + aKoff + 8] = v1;
#endif
      if (kb + 32 < K) __builtin_prefetch(gp + 32, 0, 1);
    }
    // stage B tile transposed so k-pairs are contiguous for fragments
    {
      const u16* gp = Bm + (size_t)(kb + bK) * N + nBase + bN;
      uint4 v0 = *(const uint4*)gp;
      uint4 v1 = *(const uint4*)(gp + 8);
      const u16* e0 = (const u16*)&v0;
      const u16* e1 = (const u16*)&v1;
      #pragma unroll
      for (int j = 0; j < 8; ++j) {
        ldsB[(bN + j)     * 40 + bK] = e0[j];
        ldsB[(bN + 8 + j) * 40 + bK] = e1[j];
      }
      if (kb + 32 < K) __builtin_prefetch(gp + (size_t)32 * N, 0, 1);
    }
#if USE_ASYNC_LDS
    wait_async0();
#endif
    __syncthreads();

    FragB af[2], bfr[4];
    #pragma unroll
    for (int mi = 0; mi < 2; ++mi) {
      int row = waveM * 32 + mi * 16 + ln;
      #pragma unroll
      for (int i = 0; i < 4; ++i)
        af[mi].u[i] = *(const u32*)&ldsA[row * 40 + hi * 8 + 2 * i];
      #pragma unroll
      for (int i = 0; i < 4; ++i)
        af[mi].u[4 + i] = *(const u32*)&ldsA[row * 40 + 16 + hi * 8 + 2 * i];
    }
    #pragma unroll
    for (int ni = 0; ni < 4; ++ni) {
      int n = waveN * 64 + ni * 16 + ln;
      #pragma unroll
      for (int i = 0; i < 8; ++i)
        bfr[ni].u[i] = *(const u32*)&ldsB[n * 40 + (i >> 2) * 16 + hi * 8 + 2 * (i & 3)];
    }
    #pragma unroll
    for (int mi = 0; mi < 2; ++mi)
      #pragma unroll
      for (int ni = 0; ni < 4; ++ni)
        acc[mi][ni].v = __builtin_amdgcn_wmma_f32_16x16x32_bf16(
            false, af[mi].v, false, bfr[ni].v, (short)0, acc[mi][ni].v, false, false);
    __syncthreads();
  }

  // epilogue
  #pragma unroll
  for (int mi = 0; mi < 2; ++mi) {
    #pragma unroll
    for (int ni = 0; ni < 4; ++ni) {
      int col = nBase + waveN * 64 + ni * 16 + ln;
      float bv = bias ? bias[col] : 0.f;
      #pragma unroll
      for (int r = 0; r < 8; ++r) {
        int row = mBase + waveM * 32 + mi * 16 + r + 8 * hi;
        size_t idx = (size_t)row * N + col;
        float v = acc[mi][ni].f[r] + bv;
        if (relu)  v = v > 0.f ? v : 0.f;
        if (resid) v += resid[idx];
        if (outF) outF[idx] = v;
        if (outB) outB[idx] = f2bf(v);
      }
    }
  }
}

// ---------------- flash attention (per b,h; 16 queries per wave, 32-key steps) ----------------
__global__ __launch_bounds__(128) void k_attn(
    const u16* __restrict__ qb, const u16* __restrict__ kbuf, const u16* __restrict__ vbuf,
    const int* __restrict__ mask, u16* __restrict__ concat)
{
  __shared__ __align__(16) u16 ldsK[32 * 64];     // [key][dk]
  __shared__ __align__(16) u16 ldsVt[64 * 32];    // [dk][key] (transposed)
  __shared__ __align__(16) u16 ldsP[4 * 16 * 32]; // per-wave P scratch

  const int t    = threadIdx.x;
  const int lane = t & 31;
  const int hi   = lane >> 4;
  const int ln   = lane & 15;
  const int wave = t >> 5;
  const int b = blockIdx.z, h = blockIdx.y;
  const int qBase = blockIdx.x * 64 + wave * 16;

  // Q fragments (A layout, 16x32 per k-step), loaded once from global
  FragB qf[2];
  {
    const u16* qrow = qb + (size_t)(b * S + qBase + ln) * D + h * DK;
    #pragma unroll
    for (int ks = 0; ks < 2; ++ks) {
      #pragma unroll
      for (int i = 0; i < 4; ++i)
        qf[ks].u[i] = *(const u32*)(qrow + ks * 32 + hi * 8 + 2 * i);
      #pragma unroll
      for (int i = 0; i < 4; ++i)
        qf[ks].u[4 + i] = *(const u32*)(qrow + ks * 32 + 16 + hi * 8 + 2 * i);
    }
  }

  FragC accO[4];
  #pragma unroll
  for (int ni = 0; ni < 4; ++ni)
    #pragma unroll
    for (int r = 0; r < 8; ++r) accO[ni].f[r] = 0.f;
  float rm[8], rl[8];
  #pragma unroll
  for (int r = 0; r < 8; ++r) { rm[r] = -1e30f; rl[r] = 0.f; }

  const int sKey = t >> 2;           // 0..31
  const int sCk  = (t & 3) * 16;     // 0..48

  for (int it = 0; it < S / 32; ++it) {
    const int keyBase = it * 32;
    __syncthreads();
    // stage K (row-major, async when available) and V (transposed, VGPR path)
    {
      const u16* gk = kbuf + (size_t)(b * S + keyBase + sKey) * D + h * DK + sCk;
#if USE_ASYNC_LDS
      async_cp16(&ldsK[sKey * 64 + sCk],     gk);
      async_cp16(&ldsK[sKey * 64 + sCk + 8], gk + 8);
#else
      uint4 k0 = *(const uint4*)gk;
      uint4 k1 = *(const uint4*)(gk + 8);
      *(uint4*)&ldsK[sKey * 64 + sCk]     = k0;
      *(uint4*)&ldsK[sKey * 64 + sCk + 8] = k1;
#endif
      const u16* gv = vbuf + (size_t)(b * S + keyBase + sKey) * D + h * DK + sCk;
      uint4 w0 = *(const uint4*)gv;
      uint4 w1 = *(const uint4*)(gv + 8);
      const u16* e0 = (const u16*)&w0;
      const u16* e1 = (const u16*)&w1;
      #pragma unroll
      for (int j = 0; j < 8; ++j) {
        ldsVt[(sCk + j)     * 32 + sKey] = e0[j];
        ldsVt[(sCk + 8 + j) * 32 + sKey] = e1[j];
      }
    }
#if USE_ASYNC_LDS
    wait_async0();
#endif
    __syncthreads();

    // scores: S_j(16x16) = Q(16x64) . K_j^T  -> 2 tiles x 2 k-steps = 4 WMMAs
    FragC sc[2];
    #pragma unroll
    for (int j = 0; j < 2; ++j) {
      #pragma unroll
      for (int r = 0; r < 8; ++r) sc[j].f[r] = 0.f;
      #pragma unroll
      for (int ks = 0; ks < 2; ++ks) {
        FragB kf;
        #pragma unroll
        for (int i = 0; i < 8; ++i)
          kf.u[i] = *(const u32*)&ldsK[(j * 16 + ln) * 64 + ks * 32 + (i >> 2) * 16 + hi * 8 + 2 * (i & 3)];
        sc[j].v = __builtin_amdgcn_wmma_f32_16x16x32_bf16(
            false, qf[ks].v, false, kf.v, (short)0, sc[j].v, false, false);
      }
    }

    // online softmax (rows live per 16-lane half in C layout)
    float pm0[8], pm1[8];
    #pragma unroll
    for (int r = 0; r < 8; ++r) {
      float s0 = sc[0].f[r] * 0.125f;
      float s1 = sc[1].f[r] * 0.125f;
      int qrow = qBase + r + 8 * hi;
      const int* mrow = mask + ((size_t)b * S + qrow) * S + keyBase;
      if (mrow[ln] == 0)      s0 = -1e9f;
      if (mrow[16 + ln] == 0) s1 = -1e9f;
      float tm = fmaxf(s0, s1);
      #pragma unroll
      for (int m = 1; m <= 8; m <<= 1) tm = fmaxf(tm, __shfl_xor(tm, m, 32));
      float nm = fmaxf(rm[r], tm);
      float sclr = __expf(rm[r] - nm);
      float p0 = __expf(s0 - nm);
      float p1 = __expf(s1 - nm);
      float ps = p0 + p1;
      #pragma unroll
      for (int m = 1; m <= 8; m <<= 1) ps += __shfl_xor(ps, m, 32);
      rl[r] = rl[r] * sclr + ps;
      rm[r] = nm;
      #pragma unroll
      for (int ni = 0; ni < 4; ++ni) accO[ni].f[r] *= sclr;
      pm0[r] = p0; pm1[r] = p1;
    }

    // P: C-layout -> A-layout via LDS bounce
    u16* pp = &ldsP[wave * 512];
    #pragma unroll
    for (int r = 0; r < 8; ++r) {
      pp[(r + 8 * hi) * 32 + ln]      = f2bf(pm0[r]);
      pp[(r + 8 * hi) * 32 + 16 + ln] = f2bf(pm1[r]);
    }
    __syncthreads();

    FragB pf;
    #pragma unroll
    for (int i = 0; i < 4; ++i)
      pf.u[i] = *(const u32*)&pp[ln * 32 + hi * 8 + 2 * i];
    #pragma unroll
    for (int i = 0; i < 4; ++i)
      pf.u[4 + i] = *(const u32*)&pp[ln * 32 + 16 + hi * 8 + 2 * i];

    // O(16x64) += P(16x32) . V(32x64): 4 WMMAs
    #pragma unroll
    for (int ni = 0; ni < 4; ++ni) {
      FragB vf;
      #pragma unroll
      for (int i = 0; i < 8; ++i)
        vf.u[i] = *(const u32*)&ldsVt[(ni * 16 + ln) * 32 + (i >> 2) * 16 + hi * 8 + 2 * (i & 3)];
      accO[ni].v = __builtin_amdgcn_wmma_f32_16x16x32_bf16(
          false, pf.v, false, vf.v, (short)0, accO[ni].v, false, false);
    }
  }

  // finalize + write concat (bf16, heads already in [B,S,H*64] layout)
  #pragma unroll
  for (int r = 0; r < 8; ++r) {
    float inv = rl[r] > 0.f ? 1.f / rl[r] : 0.f;
    int row = b * S + qBase + r + 8 * hi;
    #pragma unroll
    for (int ni = 0; ni < 4; ++ni)
      concat[(size_t)row * D + h * DK + ni * 16 + ln] = f2bf(accO[ni].f[r] * inv);
  }
}

// ---------------- layernorm (torch semantics: ddof=1, eps on std) ----------------
__global__ __launch_bounds__(256) void k_layernorm(
    const float* __restrict__ x, const float* __restrict__ alpha,
    const float* __restrict__ beta, float* __restrict__ out)
{
  __shared__ float s1[256], s2[256];
  int row = blockIdx.x;
  const float* xr = x + (size_t)row * D;
  float a = 0.f, q = 0.f;
  for (int c = threadIdx.x; c < D; c += 256) { float v = xr[c]; a += v; q += v * v; }
  s1[threadIdx.x] = a; s2[threadIdx.x] = q;
  __syncthreads();
  for (int st = 128; st > 0; st >>= 1) {
    if (threadIdx.x < st) { s1[threadIdx.x] += s1[threadIdx.x + st]; s2[threadIdx.x] += s2[threadIdx.x + st]; }
    __syncthreads();
  }
  float sum = s1[0], sumsq = s2[0];
  float mu  = sum / D;
  float var = (sumsq - sum * mu) / (D - 1);
  float sd  = sqrtf(fmaxf(var, 0.f));
  float inv = 1.f / (sd + 1e-6f);
  for (int c = threadIdx.x; c < D; c += 256)
    out[(size_t)row * D + c] = alpha[c] * ((xr[c] - mu) * inv) + beta[c];
}

// ---------------- orchestration ----------------
extern "C" void kernel_launch(void* const* d_in, const int* in_sizes, int n_in,
                              void* d_out, int out_size, void* d_ws, size_t ws_size,
                              hipStream_t stream) {
  (void)in_sizes; (void)n_in; (void)out_size; (void)ws_size;
  const float* x    = (const float*)d_in[0];
  const int*   mask = (const int*)d_in[1];
  const float* wq   = (const float*)d_in[2];
  const float* bq   = (const float*)d_in[3];
  const float* wk   = (const float*)d_in[4];
  const float* bk   = (const float*)d_in[5];
  const float* wv   = (const float*)d_in[6];
  const float* bv   = (const float*)d_in[7];
  const float* wo   = (const float*)d_in[8];
  const float* bo   = (const float*)d_in[9];
  const float* w1   = (const float*)d_in[10];
  const float* b1   = (const float*)d_in[11];
  const float* w2   = (const float*)d_in[12];
  const float* b2   = (const float*)d_in[13];
  const float* alpha = (const float*)d_in[14];
  const float* beta  = (const float*)d_in[15];

  char* ws = (char*)d_ws;
  size_t off = 0;
  auto alloc = [&](size_t bytes) -> void* {
    off = (off + 255) & ~(size_t)255;
    void* p = ws + off;
    off += bytes;
    return p;
  };

  u16* xb   = (u16*)alloc((size_t)BS * D * 2);
  u16* wqb  = (u16*)alloc((size_t)D * D * 2);
  u16* wkb  = (u16*)alloc((size_t)D * D * 2);
  u16* wvb  = (u16*)alloc((size_t)D * D * 2);
  u16* wob  = (u16*)alloc((size_t)D * D * 2);
  u16* w1b  = (u16*)alloc((size_t)D * DFF * 2);
  u16* w2b  = (u16*)alloc((size_t)DFF * D * 2);
  u16* qbuf = (u16*)alloc((size_t)BS * D * 2);
  u16* kbb  = (u16*)alloc((size_t)BS * D * 2);
  u16* vbb  = (u16*)alloc((size_t)BS * D * 2);
  u16* ccat = (u16*)alloc((size_t)BS * D * 2);
  float* y  = (float*)alloc((size_t)BS * D * 4);
  u16* yb   = (u16*)alloc((size_t)BS * D * 2);
  u16* h1   = (u16*)alloc((size_t)BS * DFF * 2);
  float* x2 = (float*)alloc((size_t)BS * D * 4);

  // convert / pack to bf16
  {
    int n = BS * D;
    k_cvt_bf16<<<(n + 255) / 256, 256, 0, stream>>>(x, xb, n);
  }
  k_pack_qkvw<<<(D * D) / 256, 256, 0, stream>>>(wq, wqb);
  k_pack_qkvw<<<(D * D) / 256, 256, 0, stream>>>(wk, wkb);
  k_pack_qkvw<<<(D * D) / 256, 256, 0, stream>>>(wv, wvb);
  k_cvt_bf16<<<(D * D) / 256, 256, 0, stream>>>(wo, wob, D * D);
  k_cvt_bf16<<<(D * DFF) / 256, 256, 0, stream>>>(w1, w1b, D * DFF);
  k_cvt_bf16<<<(DFF * D) / 256, 256, 0, stream>>>(w2, w2b, DFF * D);

  // QKV projections (bias flattened [H,Dk] == [h*64+k])
  k_gemm<<<dim3(D / 128, BS / 128), 256, 0, stream>>>(xb, wqb, bq, nullptr, nullptr, qbuf, BS, D, D, 0);
  k_gemm<<<dim3(D / 128, BS / 128), 256, 0, stream>>>(xb, wkb, bk, nullptr, nullptr, kbb, BS, D, D, 0);
  k_gemm<<<dim3(D / 128, BS / 128), 256, 0, stream>>>(xb, wvb, bv, nullptr, nullptr, vbb, BS, D, D, 0);

  // attention -> concat (bf16)
  k_attn<<<dim3(S / 64, H, Bb), 128, 0, stream>>>(qbuf, kbb, vbb, mask, ccat);

  // out-proj + residual(x): y (f32) and yb (bf16)
  k_gemm<<<dim3(D / 128, BS / 128), 256, 0, stream>>>(ccat, wob, bo, x, y, yb, BS, D, D, 0);

  // FFN
  k_gemm<<<dim3(DFF / 128, BS / 128), 256, 0, stream>>>(yb, w1b, b1, nullptr, nullptr, h1, BS, DFF, D, 1);
  k_gemm<<<dim3(D / 128, BS / 128), 256, 0, stream>>>(h1, w2b, b2, y, x2, nullptr, BS, D, DFF, 0);

  // layernorm -> output
  k_layernorm<<<BS, 256, 0, stream>>>(x2, alpha, beta, (float*)d_out);
}
